// TransformerBlock_69303592288777
// MI455X (gfx1250) — compile-verified, matrix operand order checked by reading, not run on hardware
//
#include <hip/hip_runtime.h>
#include <hip/hip_bf16.h>
#include <math.h>

typedef __attribute__((ext_vector_type(16))) _Float16 v16h;
typedef __attribute__((ext_vector_type(8)))  _Float16 v8h;
typedef __attribute__((ext_vector_type(8)))  float    v8f;
typedef __attribute__((ext_vector_type(4)))  unsigned u32x4;
typedef __attribute__((ext_vector_type(8)))  unsigned u32x8;

// ---------------------------------------------------------------------------
// WMMA helpers (gfx1250, wave32).
// A-fragment (16x32 f16): lane m = lane&15, hi = lane>>4:
//   elements 0..7  = row m, k = 8*hi + 0..7
//   elements 8..15 = row m, k = 16 + 8*hi + 0..7
// -> two contiguous 16B LDS loads per lane. B fragments use the same loader
// on B^T (row = output column n). C/D: element r -> row r+8*hi, col lane&15.
// ---------------------------------------------------------------------------
__device__ inline v16h load_frag(const _Float16* base, int ld) {
    int lane = threadIdx.x & 31;
    int m    = lane & 15;
    int hi   = lane >> 4;
    const _Float16* p = base + m * ld + 8 * hi;
    union { v16h v; v8h h[2]; } u;
    u.h[0] = *(const v8h*)(p);
    u.h[1] = *(const v8h*)(p + 16);
    return u.v;
}

__device__ inline v8f wmma_f16(v16h a, v16h b, v8f c) {
    return __builtin_amdgcn_wmma_f32_16x16x32_f16(false, a, false, b,
                                                  (short)0, c, false, false);
}

// ---------------------------------------------------------------------------
// Tensor Data Mover: 2-D f16 tile load, global -> LDS, with LDS padding so a
// row of tile_d0 halfs lands on a (tile_d0 + pad) stride.  D# per ISA ch.8:
//   group0: count=1 | lds_addr | global_addr[56:0] | type=2
//   group1: data_size=1(2B), pad_enable, pad_interval/amount, tensor/tile
//           dims (tensor dims := tile dims; tiles are always in-bounds here),
//           row stride (elements).  Groups 2/3 zero (dims 2+ unused).
// Issued per-wave (EXEC ignored); completion via TENSORcnt (in-order).
// ---------------------------------------------------------------------------
__device__ inline void tdm_load_2d(unsigned lds_byte_off, const void* gptr,
                                   unsigned tile_d0, unsigned tile_d1,
                                   unsigned long long row_stride_elems,
                                   unsigned pad_interval, unsigned pad_amount) {
    unsigned long long ga = (unsigned long long)(uintptr_t)gptr;
    u32x4 g0;
    g0[0] = 1u;                                   // count=1, user mode
    g0[1] = lds_byte_off;
    g0[2] = (unsigned)ga;
    g0[3] = ((unsigned)(ga >> 32) & 0x01FFFFFFu) | 0x80000000u;  // type=2
    u32x8 g1;
    g1[0] = (1u << 16) | (1u << 20) | (pad_interval << 22) | (pad_amount << 25);
    g1[1] = (tile_d0 & 0xFFFFu) << 16;                         // tensor_dim0 lo
    g1[2] = (tile_d0 >> 16) | ((tile_d1 & 0xFFFFu) << 16);     // d0 hi | d1 lo
    g1[3] = (tile_d1 >> 16) | (tile_d0 << 16);                 // d1 hi | tile_dim0
    g1[4] = tile_d1 & 0xFFFFu;                                 // tile_dim1 (dim2=0)
    g1[5] = (unsigned)row_stride_elems;                        // dim0_stride lo
    g1[6] = (unsigned)(row_stride_elems >> 32) & 0xFFFFu;      // hi | dim1_stride=0
    g1[7] = 0u;
    u32x4 gz = {0u, 0u, 0u, 0u};
    asm volatile("tensor_load_to_lds %0, %1, %2, %3"
                 :: "s"(g0), "s"(g1), "s"(gz), "s"(gz) : "memory");
}

// ---------------------------------------------------------------------------
// LayerNorm: one wave per row of length Cn, f16 output.
// ---------------------------------------------------------------------------
__global__ __launch_bounds__(256) void ln_kernel(
    const float* __restrict__ x, const float* __restrict__ g,
    const float* __restrict__ b, _Float16* __restrict__ out, int Cn) {
    int row  = blockIdx.x * 8 + (threadIdx.x >> 5);
    int lane = threadIdx.x & 31;
    const float* xr = x + (size_t)row * Cn;
    float s = 0.f, ss = 0.f;
    for (int c = lane; c < Cn; c += 32) { float v = xr[c]; s += v; ss += v * v; }
    for (int m = 16; m; m >>= 1) {
        s  += __shfl_xor(s,  m, 32);
        ss += __shfl_xor(ss, m, 32);
    }
    float mean = s / Cn;
    float var  = ss / Cn - mean * mean;
    float inv  = rsqrtf(var + 1e-5f);
    _Float16* o = out + (size_t)row * Cn;
    for (int c = lane; c < Cn; c += 32)
        o[c] = (_Float16)((xr[c] - mean) * inv * g[c] + b[c]);
}

// ---------------------------------------------------------------------------
// fp32 [R,Cc] -> f16 transposed [Cc,R]  (32x32 LDS tiles, 256 threads)
// ---------------------------------------------------------------------------
__global__ __launch_bounds__(256) void transpose_cvt_kernel(
    const float* __restrict__ in, _Float16* __restrict__ out, int R, int Cc) {
    __shared__ float tile[32][33];
    int r0 = blockIdx.y * 32, c0 = blockIdx.x * 32;
    int tx = threadIdx.x & 31, ty = threadIdx.x >> 5;
    for (int i = ty; i < 32; i += 8)
        tile[i][tx] = in[(size_t)(r0 + i) * Cc + c0 + tx];
    __syncthreads();
    for (int i = ty; i < 32; i += 8)
        out[(size_t)(c0 + i) * R + r0 + tx] = (_Float16)tile[tx][i];
}

// ---------------------------------------------------------------------------
// WMMA GEMM: C[M,N] = A[M,K](f16,row) * W[N,K](f16, pre-transposed)
// (+bias)(+GELU)(+resid).  Block 128x128, BK=32, 8 waves in a 4x2 grid, each
// wave owns a 32x64 tile: per k-step 6 fragment loads feed 8 WMMAs.
// Both tiles fetched by the TDM (pad -> LDS stride 40 halfs), double-buffered:
// wave0 issues the next tile's DMAs before computing on the current one (TDM
// completes in-order per wave -> s_wait_tensorcnt 2 releases current tile).
// ---------------------------------------------------------------------------
template <int ACT>
__global__ __launch_bounds__(256) void gemm_kernel(
    const _Float16* __restrict__ A, const _Float16* __restrict__ Wt,
    const float* __restrict__ bias, const float* __restrict__ resid,
    float* __restrict__ outf, _Float16* __restrict__ outh,
    int M, int N, int K) {
    constexpr int BM = 128, BN = 128, BK = 32, LDT = 40;
    constexpr int A_HALFS   = BM * LDT;            // 5120
    constexpr int B_HALFS   = BN * LDT;            // 5120
    constexpr int BUF_HALFS = A_HALFS + B_HALFS;   // 10240
    __shared__ alignas(16) _Float16 smem[2 * BUF_HALFS];

    int m0 = blockIdx.y * BM, n0 = blockIdx.x * BN;
    int tid = threadIdx.x, wave = tid >> 5, lane = tid & 31;
    int wm = (wave & 3) * 32, wn = (wave >> 2) * 64;

    v8f acc[2][4] = {};

    if (wave == 0) {
        tdm_load_2d(0u, A + (size_t)m0 * K, BK, BM, (unsigned long long)K, 3, 3);
        tdm_load_2d((unsigned)(A_HALFS * 2), Wt + (size_t)n0 * K, BK, BN,
                    (unsigned long long)K, 3, 3);
    }

    int buf = 0;
    for (int kb = 0; kb < K; kb += BK) {
        if (wave == 0) {
            if (kb + BK < K) {
                unsigned base = (unsigned)((buf ^ 1) * BUF_HALFS * 2);
                tdm_load_2d(base, A + (size_t)m0 * K + kb + BK, BK, BM,
                            (unsigned long long)K, 3, 3);
                tdm_load_2d(base + (unsigned)(A_HALFS * 2),
                            Wt + (size_t)n0 * K + kb + BK, BK, BN,
                            (unsigned long long)K, 3, 3);
                __builtin_amdgcn_s_wait_tensorcnt(2);   // current tile done
            } else {
                __builtin_amdgcn_s_wait_tensorcnt(0);
            }
        }
        __syncthreads();

        const _Float16* As = smem + buf * BUF_HALFS;
        const _Float16* Bt = As + A_HALFS;
        v16h af[2], bf[4];
#pragma unroll
        for (int mi = 0; mi < 2; ++mi)
            af[mi] = load_frag(&As[(wm + 16 * mi) * LDT], LDT);
#pragma unroll
        for (int ni = 0; ni < 4; ++ni)
            bf[ni] = load_frag(&Bt[(wn + 16 * ni) * LDT], LDT);
#pragma unroll
        for (int mi = 0; mi < 2; ++mi)
#pragma unroll
            for (int ni = 0; ni < 4; ++ni)
                acc[mi][ni] = wmma_f16(af[mi], bf[ni], acc[mi][ni]);
        __syncthreads();
        buf ^= 1;
    }

    int hi = lane >> 4, nl = lane & 15;
#pragma unroll
    for (int mi = 0; mi < 2; ++mi)
#pragma unroll
        for (int ni = 0; ni < 4; ++ni) {
            int col  = n0 + wn + 16 * ni + nl;
            float bv = bias ? bias[col] : 0.f;
#pragma unroll
            for (int r = 0; r < 8; ++r) {
                int row = m0 + wm + 16 * mi + r + 8 * hi;
                float v = acc[mi][ni][r] + bv;
                if (ACT == 1) v = 0.5f * v * (1.0f + erff(v * 0.70710678118654752f));
                size_t idx = (size_t)row * N + col;
                if (resid) v += resid[idx];
                if (outf) outf[idx] = v;
                if (outh) outh[idx] = (_Float16)v;
            }
        }
}

// ---------------------------------------------------------------------------
// Flash attention (causal). qkv: [B*T, 3C] f16 (q|k|v). One block = one
// (b,h) x 64 query rows; 4 waves x 16 rows; kv streamed in 32-row tiles.
// K tile fetched by TDM (row stride 3C, 128B rows padded +16B -> stride 72
// halfs) overlapped with the manual V-transpose staging.
// ---------------------------------------------------------------------------
__global__ __launch_bounds__(128) void attn_kernel(
    const _Float16* __restrict__ qkv, _Float16* __restrict__ out,
    int T, int Cn, int H) {
    constexpr int D = 64;
    constexpr int OFF_Q = 0;                 // 64*72 halfs
    constexpr int OFF_K = 64 * 72;           // 32*72 halfs
    constexpr int OFF_V = OFF_K + 32 * 72;   // Vt[d][kk], 64*40 halfs
    constexpr int OFF_P = OFF_V + 64 * 40;   // per-wave 16*40 halfs
    __shared__ alignas(16) _Float16 smem[OFF_P + 4 * 16 * 40];

    int bh = blockIdx.y;
    int b  = bh / H, h = bh % H;
    int q0 = blockIdx.x * 64;
    int tid = threadIdx.x, wave = tid >> 5, lane = tid & 31;
    int hi = lane >> 4, nl = lane & 15;
    const int C3 = 3 * Cn;

    const _Float16* qbase = qkv + (size_t)(b * T) * C3 + h * D;

    // Stage Q (scaled by 1/sqrt(D) = 0.125).
    for (int i = tid; i < 64 * 64; i += 128) {
        int r = i >> 6, d = i & 63;
        float qv = (float)qbase[(size_t)(q0 + r) * C3 + d];
        smem[OFF_Q + r * 72 + d] = (_Float16)(qv * 0.125f);
    }
    __syncthreads();

    v16h aq0 = load_frag(&smem[OFF_Q + wave * 16 * 72], 72);       // d 0..31
    v16h aq1 = load_frag(&smem[OFF_Q + wave * 16 * 72 + 32], 72);  // d 32..63

    v8f o_[4] = {};
    float mr[8], lr[8];
#pragma unroll
    for (int r = 0; r < 8; ++r) { mr[r] = -1e30f; lr[r] = 0.f; }

    int kend = q0 + 64;  // causal upper bound for this q tile
    for (int kb = 0; kb < kend; kb += 32) {
        // TDM: K tile (32 rows x 64 halfs, stride 3C) -> LDS stride 72.
        if (wave == 0)
            tdm_load_2d((unsigned)(OFF_K * 2), qbase + (size_t)kb * C3 + Cn,
                        64, 32, (unsigned long long)C3, 4, 3);
        // Meanwhile: stage V transposed (Vt[d][kk]).
        for (int i = tid; i < 32 * 64; i += 128) {
            int kk = i >> 6, d = i & 63;
            smem[OFF_V + d * 40 + kk] = qbase[(size_t)(kb + kk) * C3 + 2 * Cn + d];
        }
        if (wave == 0) __builtin_amdgcn_s_wait_tensorcnt(0);
        __syncthreads();

        // S = Q K^T  (two 16x16 tiles along kk, 2 WMMAs each over d)
        v8f s[2];
#pragma unroll
        for (int j = 0; j < 2; ++j) {
            v8f z = {};
            v16h b0 = load_frag(&smem[OFF_K + (16 * j) * 72], 72);
            v16h b1 = load_frag(&smem[OFF_K + (16 * j) * 72 + 32], 72);
            z = wmma_f16(aq0, b0, z);
            z = wmma_f16(aq1, b1, z);
            s[j] = z;
        }

        // Causal mask.
#pragma unroll
        for (int j = 0; j < 2; ++j) {
            int col = kb + 16 * j + nl;
#pragma unroll
            for (int r = 0; r < 8; ++r) {
                int row = q0 + wave * 16 + r + 8 * hi;
                if (col > row) s[j][r] = -1e30f;
            }
        }

        // Online softmax (row r+8*hi spread over a 16-lane group).
#pragma unroll
        for (int r = 0; r < 8; ++r) {
            float rm = fmaxf(s[0][r], s[1][r]);
            for (int msk = 1; msk < 16; msk <<= 1)
                rm = fmaxf(rm, __shfl_xor(rm, msk, 32));
            float mnew  = fmaxf(mr[r], rm);
            float alpha = __expf(mr[r] - mnew);
            float p0 = __expf(s[0][r] - mnew);
            float p1 = __expf(s[1][r] - mnew);
            s[0][r] = p0; s[1][r] = p1;
            float rs = p0 + p1;
            for (int msk = 1; msk < 16; msk <<= 1)
                rs += __shfl_xor(rs, msk, 32);
            lr[r] = lr[r] * alpha + rs;
            mr[r] = mnew;
#pragma unroll
            for (int jd = 0; jd < 4; ++jd) o_[jd][r] = o_[jd][r] * alpha;
        }

        // P (D-layout) -> LDS -> A-fragment.
        _Float16* pw = &smem[OFF_P + wave * 16 * 40];
#pragma unroll
        for (int j = 0; j < 2; ++j)
#pragma unroll
            for (int r = 0; r < 8; ++r)
                pw[(r + 8 * hi) * 40 + 16 * j + nl] = (_Float16)s[j][r];
        asm volatile("s_wait_dscnt 0" ::: "memory");

        v16h ap = load_frag(pw, 40);
#pragma unroll
        for (int jd = 0; jd < 4; ++jd) {
            v16h bv = load_frag(&smem[OFF_V + (16 * jd) * 40], 40);
            o_[jd] = wmma_f16(ap, bv, o_[jd]);
        }
        __syncthreads();
    }

    // Normalize + write out (f16).
#pragma unroll
    for (int jd = 0; jd < 4; ++jd)
#pragma unroll
        for (int r = 0; r < 8; ++r) {
            int row = q0 + wave * 16 + r + 8 * hi;
            int col = h * D + 16 * jd + nl;
            out[(size_t)(b * T + row) * Cn + col] = (_Float16)(o_[jd][r] / lr[r]);
        }
}

// ---------------------------------------------------------------------------
// Launch
// ---------------------------------------------------------------------------
extern "C" void kernel_launch(void* const* d_in, const int* in_sizes, int n_in,
                              void* d_out, int out_size, void* d_ws, size_t ws_size,
                              hipStream_t stream) {
    (void)in_sizes; (void)n_in; (void)out_size; (void)ws_size;
    const float* x     = (const float*)d_in[0];
    // d_in[1] = causal_mask (bool) -- handled analytically
    const float* qkv_w = (const float*)d_in[2];
    const float* out_w = (const float*)d_in[3];
    const float* out_b = (const float*)d_in[4];
    const float* ln1_g = (const float*)d_in[5];
    const float* ln1_b = (const float*)d_in[6];
    const float* ln2_g = (const float*)d_in[7];
    const float* ln2_b = (const float*)d_in[8];
    const float* ff_w1 = (const float*)d_in[9];
    const float* ff_b1 = (const float*)d_in[10];
    const float* ff_w2 = (const float*)d_in[11];
    const float* ff_b2 = (const float*)d_in[12];
    float* out = (float*)d_out;

    constexpr int Bc = 4, Tc = 2048, Cc = 1024, Hc = 16, FFc = 4096;
    constexpr size_t BT = (size_t)Bc * Tc;

    char* ws = (char*)d_ws;
    size_t off = 0;
    auto alloc = [&](size_t n) {
        char* p = ws + off;
        off = (off + n + 255) & ~(size_t)255;
        return p;
    };
    // Weights f16, pre-transposed to [N,K] for the TDM-fed GEMM.
    _Float16* wqT   = (_Float16*)alloc((size_t)3 * Cc * Cc * 2);   // [3C, C]
    _Float16* woT   = (_Float16*)alloc((size_t)Cc * Cc * 2);       // [C, C]
    _Float16* w1T   = (_Float16*)alloc((size_t)FFc * Cc * 2);      // [FF, C]
    _Float16* w2T   = (_Float16*)alloc((size_t)Cc * FFc * 2);      // [C, FF]
    _Float16* xln1  = (_Float16*)alloc(BT * Cc * 2);
    _Float16* qkvh  = (_Float16*)alloc(BT * 3 * Cc * 2);
    _Float16* attnh = (_Float16*)alloc(BT * Cc * 2);
    float*    x1    = (float*)   alloc(BT * Cc * 4);
    _Float16* h2    = (_Float16*)alloc(BT * Cc * 2);
    _Float16* ff1   = (_Float16*)alloc(BT * FFc * 2);

    auto tcvt = [&](const float* src, _Float16* dst, int R, int Ccols) {
        transpose_cvt_kernel<<<dim3(Ccols / 32, R / 32), 256, 0, stream>>>(
            src, dst, R, Ccols);
    };
    tcvt(qkv_w, wqT, Cc, 3 * Cc);
    tcvt(out_w, woT, Cc, Cc);
    tcvt(ff_w1, w1T, Cc, FFc);
    tcvt(ff_w2, w2T, FFc, Cc);

    // ln1(x) -> f16
    ln_kernel<<<BT / 8, 256, 0, stream>>>(x, ln1_g, ln1_b, xln1, Cc);

    // qkv = ln1(x) @ qkv_w   (f16 out)
    gemm_kernel<0><<<dim3(3 * Cc / 128, BT / 128), 256, 0, stream>>>(
        xln1, wqT, nullptr, nullptr, nullptr, qkvh, (int)BT, 3 * Cc, Cc);

    // causal flash attention -> f16
    attn_kernel<<<dim3(Tc / 64, Bc * Hc), 128, 0, stream>>>(qkvh, attnh, Tc, Cc, Hc);

    // x1 = x + attn @ out_w + out_b   (f32)
    gemm_kernel<0><<<dim3(Cc / 128, BT / 128), 256, 0, stream>>>(
        attnh, woT, out_b, x, x1, nullptr, (int)BT, Cc, Cc);

    // h2 = ln2(x1) -> f16
    ln_kernel<<<BT / 8, 256, 0, stream>>>(x1, ln2_g, ln2_b, h2, Cc);

    // ff1 = gelu(h2 @ ff_w1 + ff_b1) -> f16
    gemm_kernel<1><<<dim3(FFc / 128, BT / 128), 256, 0, stream>>>(
        h2, w1T, ff_b1, nullptr, nullptr, ff1, (int)BT, FFc, Cc);

    // out = x1 + ff1 @ ff_w2 + ff_b2  (f32)
    gemm_kernel<0><<<dim3(Cc / 128, BT / 128), 256, 0, stream>>>(
        ff1, w2T, ff_b2, x1, out, nullptr, (int)BT, Cc, FFc);
}
